// Model_3212635537914
// MI455X (gfx1250) — matrix operand order repaired and checked by reading, not compiled, not run on hardware
//
#include <hip/hip_runtime.h>

// ---------------------------------------------------------------------------
// BiLSTM-CRF for MI455X (gfx1250, wave32).
//   K0a: embedding gather + f32->bf16 (padded K=300->320)
//   K0b: pack Wih^T into WMMA B-fragment layout (bf16)
//   K0c: init h buffers + barrier counters
//   K1 : batched input projection GEMM via v_wmma_f32_16x16x32_bf16,
//        2 accumulator chains per wave (M=32) to hide WMMA->WMMA RAW hazard
//   K2 : sequential LSTM recurrence, Whh resident in LDS, 4 WGPs/direction,
//        cross-WGP step barrier via L2 atomics (agent scope)
//   K3 : Viterbi decode, single wave, backpointers in LDS
// ---------------------------------------------------------------------------

#define T_LEN 4096
#define E_DIM 300
#define KPAD  320
#define HOUT  1024   // 4*H2
#define H2    256
#define KTAG  32
#define START_TAG 30
#define STOP_TAG  31
#define NEGV (-10000.0f)

typedef __attribute__((ext_vector_type(16))) __bf16        v16bf;
typedef __attribute__((ext_vector_type(8)))  float         v8f;
typedef __attribute__((ext_vector_type(4)))  unsigned int  v4u;

__device__ __forceinline__ unsigned short f2bf(float f) {
  unsigned int u = __builtin_bit_cast(unsigned int, f);
  unsigned int r = u + 0x7FFFu + ((u >> 16) & 1u);   // round-to-nearest-even
  return (unsigned short)(r >> 16);
}

__device__ __forceinline__ float sigf(float x) { return 1.0f / (1.0f + expf(-x)); }

// --------------------------- K0a: gather + convert -------------------------
__global__ __launch_bounds__(KPAD)
void k_gather(const int* __restrict__ sentence, const float* __restrict__ embed,
              unsigned short* __restrict__ Xbf) {
  const int t = blockIdx.x;
  const int k = threadIdx.x;
  unsigned short v = 0;
  if (k < E_DIM) v = f2bf(embed[(long long)sentence[t] * E_DIM + k]);
  Xbf[t * KPAD + k] = v;
}

// --------------------------- K0b: pack B fragments -------------------------
// B-fragment order used here (contiguous k, same as A; 16x16x32 bf16): lane l holds column N = l&15,
// K = 16*(l>>4) + e for element e in 0..15.  Fragment for k-tile kt, n-tile nt
// stored contiguously: off = ((kt*64 + nt)*32 + l)*16 + e.
__global__ __launch_bounds__(256)
void k_packW(const float* __restrict__ Wih_f, const float* __restrict__ Wih_b,
             unsigned short* __restrict__ Bpack) {
  const int idx = blockIdx.x * 256 + threadIdx.x;       // 0 .. 655359
  const int dir = idx / 327680;
  const int r   = idx % 327680;
  const int e  = r & 15;
  const int l  = (r >> 4) & 31;
  const int nt = (r >> 9) & 63;
  const int kt = r >> 15;                                // 0..9
  const int k = kt * 32 + 16 * (l >> 4) + e;
  const int n = nt * 16 + (l & 15);
  const float* W = dir ? Wih_b : Wih_f;                  // (1024, 300) row-major
  unsigned short v = 0;
  if (k < E_DIM) v = f2bf(W[n * E_DIM + k]);
  Bpack[idx] = v;
}

// --------------------------- K0c: init state -------------------------------
__global__ __launch_bounds__(256)
void k_init(const float* __restrict__ h0, float* __restrict__ hbuf, int* __restrict__ cnt) {
  const int dir = blockIdx.x;                // 0..1
  const int n   = threadIdx.x;               // 0..255
  hbuf[dir * 512 + 0 * 256 + n] = h0[dir * 256 + n];   // buffer 0 = h at step -1
  if (n == 0) cnt[dir] = 0;
}

// --------------------------- K1: input projection GEMM ---------------------
// Z[d][t][j] = b_d[j] + sum_k Wih_d[j,k] * x[t,k]   (K padded to 320, bf16)
// Each wave owns a 32(M=t) x 16(N=j) output: two independent accumulator
// chains share one B fragment -> WMMA->WMMA accumulator RAW bubbles are
// filled by the other chain's WMMA, and B load bandwidth is halved.
__global__ __launch_bounds__(128)
void k_gemm(const unsigned short* __restrict__ Xbf, const unsigned short* __restrict__ Bpack,
            const float* __restrict__ b_f, const float* __restrict__ b_b,
            float* __restrict__ Zf, float* __restrict__ Zb) {
  const int dir  = blockIdx.z;
  const unsigned short* Bp = Bpack + dir * 327680;
  const float* bias = dir ? b_b : b_f;
  float* Z = dir ? Zb : Zf;

  const int wave = threadIdx.x >> 5;
  const int lane = threadIdx.x & 31;
  const int m0    = blockIdx.x * 32;               // t-tile pair base
  const int ntile = blockIdx.y * 4 + wave;         // j-tile (0..63)
  const int n0    = ntile * 16;
  const int g  = lane >> 4;
  const int mr = lane & 15;

  union Frag { v16bf v; v4u q[2]; };

  v8f acc0 = {0.f, 0.f, 0.f, 0.f, 0.f, 0.f, 0.f, 0.f};
  v8f acc1 = {0.f, 0.f, 0.f, 0.f, 0.f, 0.f, 0.f, 0.f};
  for (int kt = 0; kt < 10; ++kt) {
    Frag a0, a1, b;
    // A (16x32 bf16): lane holds row M = lane&15; K = 16*(lane>>4) + e
    const unsigned short* ap0 = Xbf + (m0 + mr) * KPAD + kt * 32 + 16 * g;
    const unsigned short* ap1 = ap0 + 16 * KPAD;
    a0.q[0] = *(const v4u*)(ap0);        // e=0..7  -> K = 16g + 0..7
    a0.q[1] = *(const v4u*)(ap0 + 8);    // e=8..15 -> K = 16g + 8..15
    a1.q[0] = *(const v4u*)(ap1);
    a1.q[1] = *(const v4u*)(ap1 + 8);
    // B: pre-packed contiguous 32B per lane (shared by both chains)
    const unsigned short* bp = Bp + (((kt * 64 + ntile) * 32 + lane) << 4);
    b.q[0] = ((const v4u*)bp)[0];
    b.q[1] = ((const v4u*)bp)[1];
    acc0 = __builtin_amdgcn_wmma_f32_16x16x32_bf16(false, a0.v, false, b.v,
                                                   (short)0, acc0, false, false);
    acc1 = __builtin_amdgcn_wmma_f32_16x16x32_bf16(false, a1.v, false, b.v,
                                                   (short)0, acc1, false, false);
  }
  // C/D layout: VGPR r -> M = r + 8*(lane>>4), N = lane&15
  const int j = n0 + mr;
  const float bs = bias[j];
#pragma unroll
  for (int r = 0; r < 8; ++r) {
    const int t = m0 + r + 8 * g;
    Z[t * HOUT + j]          = acc0[r] + bs;
    Z[(t + 16) * HOUT + j]   = acc1[r] + bs;
  }
}

// --------------------------- K2: LSTM recurrence ---------------------------
// grid = 8 blocks of 256 threads: block = dir*4 + p.  Block owns hidden slice
// n in [64p, 64p+64) -> gate rows {g*256 + 64p + nl}.  Whh slice (256x256 f32)
// lives in LDS (row pitch 257 -> conflict-free).  h exchanged via L2 each step.
__global__ __launch_bounds__(256)
void k_recur(const float* __restrict__ Zf, const float* __restrict__ Zb,
             const float* __restrict__ Whh_f, const float* __restrict__ Whh_b,
             const float* __restrict__ W_out, const float* __restrict__ c0,
             float* __restrict__ hbuf, int* __restrict__ cnt,
             float* __restrict__ partial) {
  extern __shared__ float smem[];
  float* Wsh  = smem;                 // 256 * 257
  float* hl   = Wsh + 256 * 257;      // 256   staged h
  float* zbuf = hl + 256;             // 256   gate pre-activations
  float* hnw  = zbuf + 256;           // 64    new h slice
  float* wsh  = hnw + 64;             // 32*64 W_out slice

  const int blk = blockIdx.x;
  const int dir = blk >> 2;
  const int p   = blk & 3;
  const int tid = threadIdx.x;
  const float* Z   = dir ? Zb : Zf;
  const float* Whh = dir ? Whh_b : Whh_f;

  // Load Whh slice: local row rl = g*64 + nl  ->  global j = g*256 + 64p + nl
  for (int rl = 0; rl < 256; ++rl) {
    const int gg = rl >> 6, nl = rl & 63;
    const int j = gg * 256 + p * 64 + nl;
    Wsh[rl * 257 + tid] = Whh[j * H2 + tid];
  }
  // Load W_out slice: wsh[k*64+nl] = W_out[k, dir*256 + 64p + nl]
  for (int idx = tid; idx < 32 * 64; idx += 256) {
    const int k = idx >> 6, nl = idx & 63;
    wsh[idx] = W_out[k * 512 + dir * 256 + p * 64 + nl];
  }
  float cstate = (tid < 64) ? c0[dir * 256 + p * 64 + tid] : 0.f;
  __syncthreads();

  const int g = tid >> 6, nloc = tid & 63;
  const int j = g * 256 + p * 64 + nloc;          // my gate row
  int* mycnt = cnt + dir;
  const int slab = dir * 4 + p;

  for (int s = 0; s < T_LEN; ++s) {
    const int t = dir ? (T_LEN - 1 - s) : s;
    // stage h_{s-1} (buffer s&1), coherent across WGPs
    hl[tid] = __hip_atomic_load(&hbuf[dir * 512 + (s & 1) * 256 + tid],
                                __ATOMIC_RELAXED, __HIP_MEMORY_SCOPE_AGENT);
    __syncthreads();

    float acc = Z[t * HOUT + j];
    const float* wrow = Wsh + tid * 257;
#pragma unroll 8
    for (int k = 0; k < H2; ++k) acc += wrow[k] * hl[k];
    zbuf[tid] = acc;
    __syncthreads();

    if (tid < 64) {
      const float zi = zbuf[tid], zf = zbuf[64 + tid], zg = zbuf[128 + tid], zo = zbuf[192 + tid];
      cstate = sigf(zf) * cstate + sigf(zi) * tanhf(zg);
      const float h = sigf(zo) * tanhf(cstate);
      hnw[tid] = h;
      __hip_atomic_store(&hbuf[dir * 512 + ((s + 1) & 1) * 256 + p * 64 + tid], h,
                         __ATOMIC_RELAXED, __HIP_MEMORY_SCOPE_AGENT);
    }
    __syncthreads();
    if (tid < 32) {                      // output-projection partial (deterministic)
      float pa = 0.f;
#pragma unroll 8
      for (int nl = 0; nl < 64; ++nl) pa += wsh[tid * 64 + nl] * hnw[nl];
      partial[(slab * T_LEN + t) * KTAG + tid] = pa;
    }
    __syncthreads();
    if (tid == 0) {                      // 4-WGP step barrier (monotonic counter)
      __hip_atomic_fetch_add(mycnt, 1, __ATOMIC_RELEASE, __HIP_MEMORY_SCOPE_AGENT);
      while (__hip_atomic_load(mycnt, __ATOMIC_ACQUIRE, __HIP_MEMORY_SCOPE_AGENT)
             < 4 * (s + 1)) {
        __builtin_amdgcn_s_sleep(1);
      }
    }
    __syncthreads();
  }
}

// --------------------------- K3: Viterbi decode ----------------------------
__global__ __launch_bounds__(32)
void k_viterbi(const float* __restrict__ partial, const float* __restrict__ b_out,
               const float* __restrict__ trans, float* __restrict__ out) {
  extern __shared__ float sm[];
  float* transL = sm;                          // 32*32
  float* fv     = transL + 1024;               // 32
  float* term   = fv + 32;                     // 32
  unsigned char* bp = (unsigned char*)(term + 32);  // 4096*32 backpointers

  const int i = threadIdx.x;                   // tag 0..31
  for (int idx = i; idx < 1024; idx += 32) transL[idx] = trans[idx];
  fv[i] = (i == START_TAG) ? 0.f : NEGV;
  const float bo = b_out[i];
  __syncthreads();

  for (int t = 0; t < T_LEN; ++t) {
    float feat = bo;
#pragma unroll
    for (int s = 0; s < 8; ++s) feat += partial[(s * T_LEN + t) * KTAG + i];
    float best = -3.4e38f; int arg = 0;
    for (int jj = 0; jj < KTAG; ++jj) {
      const float sc = fv[jj] + transL[i * KTAG + jj];
      if (sc > best) { best = sc; arg = jj; }
    }
    bp[t * KTAG + i] = (unsigned char)arg;
    const float nf = best + feat;
    __syncthreads();
    fv[i] = nf;
    __syncthreads();
  }
  term[i] = fv[i] + transL[STOP_TAG * KTAG + i];
  __syncthreads();
  if (i == 0) {
    float bestS = term[0]; int bestT = 0;
    for (int jj = 1; jj < KTAG; ++jj)
      if (term[jj] > bestS) { bestS = term[jj]; bestT = jj; }
    out[0] = bestS;
    int tag = bestT;
    for (int t = T_LEN - 1; t >= 0; --t) {
      out[1 + t] = (float)tag;
      tag = bp[t * KTAG + tag];
    }
  }
}

// --------------------------- host launch -----------------------------------
extern "C" void kernel_launch(void* const* d_in, const int* in_sizes, int n_in,
                              void* d_out, int out_size, void* d_ws, size_t ws_size,
                              hipStream_t stream) {
  const int*   sentence = (const int*)  d_in[0];
  const float* embed    = (const float*)d_in[1];
  const float* Wih_f    = (const float*)d_in[2];
  const float* Whh_f    = (const float*)d_in[3];
  const float* b_f      = (const float*)d_in[4];
  const float* Wih_b    = (const float*)d_in[5];
  const float* Whh_b    = (const float*)d_in[6];
  const float* b_b      = (const float*)d_in[7];
  const float* W_out    = (const float*)d_in[8];
  const float* b_out    = (const float*)d_in[9];
  const float* trans    = (const float*)d_in[10];
  const float* h0       = (const float*)d_in[11];
  const float* c0       = (const float*)d_in[12];
  float* out = (float*)d_out;

  char* ws = (char*)d_ws;
  size_t off = 0;
  unsigned short* Xbf   = (unsigned short*)(ws + off); off += (size_t)T_LEN * KPAD * 2;      // 2.5 MB
  unsigned short* Bpack = (unsigned short*)(ws + off); off += (size_t)2 * 327680 * 2;        // 1.25 MB
  float* Zf     = (float*)(ws + off); off += (size_t)T_LEN * HOUT * 4;                       // 16 MB
  float* Zb     = (float*)(ws + off); off += (size_t)T_LEN * HOUT * 4;                       // 16 MB
  float* hbuf   = (float*)(ws + off); off += 2 * 2 * 256 * 4;                                // 4 KB
  int*   cnt    = (int*)  (ws + off); off += 256;
  float* partial= (float*)(ws + off); off += (size_t)8 * T_LEN * KTAG * 4;                   // 4 MB
  (void)ws_size; (void)in_sizes; (void)n_in; (void)out_size;

  // K0: gather/convert, pack weights, init state
  k_gather<<<T_LEN, KPAD, 0, stream>>>(sentence, embed, Xbf);
  k_packW <<<(2 * 327680) / 256, 256, 0, stream>>>(Wih_f, Wih_b, Bpack);
  k_init  <<<2, 256, 0, stream>>>(h0, hbuf, cnt);

  // K1: WMMA input-projection GEMM, both directions, M=32 per wave
  k_gemm<<<dim3(T_LEN / 32, 16, 2), 128, 0, stream>>>(Xbf, Bpack, b_f, b_b, Zf, Zb);

  // K2: recurrence, 8 WGPs (4 per direction), Whh in 273 KB of LDS each
  const size_t recurLds = (256 * 257 + 256 + 256 + 64 + 32 * 64) * sizeof(float);
  k_recur<<<8, 256, recurLds, stream>>>(Zf, Zb, Whh_f, Whh_b, W_out, c0, hbuf, cnt, partial);

  // K3: Viterbi, backpointers in LDS (135 KB)
  const size_t vitLds = (1024 + 32 + 32) * sizeof(float) + (size_t)T_LEN * KTAG;
  k_viterbi<<<1, 32, vitLds, stream>>>(partial, b_out, trans, out);
}
